// GNNLayer_60997125538475
// MI455X (gfx1250) — matrix operand order repaired and checked by reading, not compiled, not run on hardware
//
#include <hip/hip_runtime.h>
#include <hip/hip_bf16.h>

// GNN layer (GINEConv + MLP + BatchNorm) for MI455X / gfx1250.
// BF16 WMMA (f32 accumulate) for all three GEMMs; fp32 epilogues.
// Round 2: 32x64 wave tiles (8 WMMA/wave/K-step), double-buffered LDS with a
// single barrier per K-step, and async-to-LDS (ASYNCcnt) staging for bf16 tiles.

typedef __attribute__((ext_vector_type(16))) __bf16 v16bf;
typedef __attribute__((ext_vector_type(8)))  __bf16 v8bf;
typedef __attribute__((ext_vector_type(8)))  float  v8f;

static constexpr int Dk     = 512;
static constexpr int NNODES = 25000;
static constexpr int NEDGES = 400000;

union FragU { v16bf v; struct { v8bf lo; v8bf hi; } p; };

__device__ __forceinline__ v16bf load_frag(const __bf16* p0, const __bf16* p1) {
    FragU u;
    u.p.lo = *(const v8bf*)p0;
    u.p.hi = *(const v8bf*)p1;
    return u.v;
}

__device__ __forceinline__ v8f wmma_bf16(v16bf a, v16bf b, v8f c) {
    return __builtin_amdgcn_wmma_f32_16x16x32_bf16(
        /*neg_a=*/false, a, /*neg_b=*/false, b,
        /*c_mod=*/(short)0, c, /*reuse_a=*/false, /*reuse_b=*/false);
}

// Async copy 16B global -> LDS (ASYNCcnt tracked). LDS address = low 32 bits of
// the flat address (LDS aperture maps addr[31:0] to the LDS offset).
__device__ __forceinline__ void async_b128(const void* g, void* l) {
    unsigned long long ga = (unsigned long long)g;
    unsigned int       la = (unsigned int)(unsigned long long)l;
    asm volatile("global_load_async_to_lds_b128 %0, %1, off"
                 :: "v"(la), "v"(ga) : "memory");
}
__device__ __forceinline__ void wait_async0() {
    asm volatile("s_wait_asynccnt 0x0" ::: "memory");
}

__device__ __forceinline__ void cvt_store8(__bf16* as, float4 f0, float4 f1) {
    as[0] = (__bf16)f0.x; as[1] = (__bf16)f0.y; as[2] = (__bf16)f0.z; as[3] = (__bf16)f0.w;
    as[4] = (__bf16)f1.x; as[5] = (__bf16)f1.y; as[6] = (__bf16)f1.z; as[7] = (__bf16)f1.w;
}

// ---------------------------------------------------------------- utilities
__global__ __launch_bounds__(256)
void zero_ws_kernel(float4* __restrict__ p) {
    long i = (long)blockIdx.x * 256 + threadIdx.x;
    p[i] = make_float4(0.f, 0.f, 0.f, 0.f);
}

// Wt[n][k] = (bf16) W[k][n]   (B-matrix column-major staging layout)
__global__ __launch_bounds__(256)
void cast_transpose_kernel(const float* __restrict__ W, __bf16* __restrict__ Wt) {
    const int k = blockIdx.x;
    for (int n = threadIdx.x; n < Dk; n += 256)
        Wt[(long)n * Dk + k] = (__bf16)W[(long)k * Dk + n];
}

// -------------------------------------------------- edge GEMM + gather/scatter
// aggr[dst] += relu(x[src] + edge_attr@W_e + b_e)
// Block tile 64(M) x 256(N), BK=32; 8 waves, each wave 32x64 (2x4 WMMA tiles).
__global__ __launch_bounds__(256)
void edge_gemm_scatter(const float* __restrict__ ea, const __bf16* __restrict__ Wt,
                       const float* __restrict__ be, const float* __restrict__ x,
                       const long long* __restrict__ eidx, float* __restrict__ aggr) {
    __shared__ __bf16 As[2][64][32];
    __shared__ __bf16 Bs[2][256][32];

    const int tid  = threadIdx.x;
    const int lane = tid & 31;
    const int wave = tid >> 5;
    const int wM   = wave & 1;   // 0..1  -> rows [wM*32, +32)
    const int wN   = wave >> 1;  // 0..3  -> cols [wN*64, +64)
    const long ebase = (long)blockIdx.x * 64;
    const int  nbase = blockIdx.y * 256;

    const int srow = tid >> 2;        // 0..63
    const int scol = (tid & 3) * 8;   // 0,8,16,24

    v8f c[2][4] = {};

    const float* gA0 = ea + (ebase + srow) * Dk + scol;

    // stage k=0 into buffer 0
    {
        float4 f0 = *(const float4*)(gA0);
        float4 f1 = *(const float4*)(gA0 + 4);
        cvt_store8(&As[0][srow][scol], f0, f1);
#pragma unroll
        for (int j = 0; j < 4; ++j) {
            const int u = tid + 256 * j;
            const int row = u >> 2, ch = (u & 3) * 8;
            async_b128(Wt + (long)(nbase + row) * Dk + ch, &Bs[0][row][ch]);
        }
    }
    wait_async0();
    __syncthreads();

    for (int i = 0; i < 16; ++i) {
        const int cur = i & 1;
        if (i < 15) {
            const int k1 = (i + 1) * 32;
            const float* gA = gA0 + k1;
            float4 f0 = *(const float4*)(gA);
            float4 f1 = *(const float4*)(gA + 4);
            __builtin_prefetch(gA + 32, 0, 1);   // global_prefetch_b8
            cvt_store8(&As[cur ^ 1][srow][scol], f0, f1);
#pragma unroll
            for (int j = 0; j < 4; ++j) {
                const int u = tid + 256 * j;
                const int row = u >> 2, ch = (u & 3) * 8;
                async_b128(Wt + (long)(nbase + row) * Dk + k1 + ch, &Bs[cur ^ 1][row][ch]);
            }
        }
        const int lr = lane & 15;
        const int ka = (lane < 16) ? 0 : 8;    // bf16 A 16x32 lane layout
        const int kb = (lane < 16) ? 0 : 8;    // bf16 B 32x16 lane layout
        v16bf a[2], b[4];
#pragma unroll
        for (int mt = 0; mt < 2; ++mt)
            a[mt] = load_frag(&As[cur][wM * 32 + mt * 16 + lr][ka],
                              &As[cur][wM * 32 + mt * 16 + lr][ka + 16]);
#pragma unroll
        for (int nt = 0; nt < 4; ++nt)
            b[nt] = load_frag(&Bs[cur][wN * 64 + nt * 16 + lr][kb],
                              &Bs[cur][wN * 64 + nt * 16 + lr][kb + 16]);
#pragma unroll
        for (int mt = 0; mt < 2; ++mt)
#pragma unroll
            for (int nt = 0; nt < 4; ++nt)
                c[mt][nt] = wmma_bf16(a[mt], b[nt], c[mt][nt]);
        wait_async0();     // after the math: prefetch overlapped with WMMA
        __syncthreads();
    }

    // epilogue: bias, gather x[src], relu, scatter-add to aggr[dst]
    const int lc      = lane & 15;
    const int colbase = nbase + wN * 64;
    float bias[4];
#pragma unroll
    for (int nt = 0; nt < 4; ++nt) bias[nt] = be[colbase + nt * 16 + lc];

#pragma unroll
    for (int mt = 0; mt < 2; ++mt) {
#pragma unroll
        for (int r = 0; r < 8; ++r) {
            const long e = ebase + wM * 32 + mt * 16 + ((lane >> 4) << 3) + r;
            const long long s = eidx[e];
            const long long d = eidx[NEDGES + e];
            const float* xs = x    + (long)s * Dk;
            float*       ag = aggr + (long)d * Dk;
#pragma unroll
            for (int nt = 0; nt < 4; ++nt) {
                const int col = colbase + nt * 16 + lc;
                float m = fmaxf(xs[col] + c[mt][nt][r] + bias[nt], 0.0f);
                atomicAdd(&ag[col], m);   // resolves in L2 (aggr = 51MB << 192MB)
            }
        }
    }
}

// ---------------------------------------------------------- node GEMM 1 (ReLU)
// H1 = bf16( relu( (x + aggr) @ W1 + b1 ) )
__global__ __launch_bounds__(256)
void node_gemm1(const float* __restrict__ x, const float* __restrict__ aggr,
                const __bf16* __restrict__ Wt, const float* __restrict__ bv,
                __bf16* __restrict__ H1) {
    __shared__ __bf16 As[2][64][32];
    __shared__ __bf16 Bs[2][256][32];

    const int tid  = threadIdx.x;
    const int lane = tid & 31;
    const int wave = tid >> 5;
    const int wM   = wave & 1;
    const int wN   = wave >> 1;
    const int mbase = blockIdx.x * 64;
    const int nbase = blockIdx.y * 256;

    const int srow = tid >> 2;
    const int scol = (tid & 3) * 8;
    const int nodeS = mbase + srow;
    const bool vrow = nodeS < NNODES;

    v8f c[2][4] = {};

    const float* gX0 = x    + (long)nodeS * Dk + scol;
    const float* gG0 = aggr + (long)nodeS * Dk + scol;

    {
        __bf16* as = &As[0][srow][scol];
        if (vrow) {
            float4 x0 = *(const float4*)(gX0);
            float4 x1 = *(const float4*)(gX0 + 4);
            float4 g0 = *(const float4*)(gG0);
            float4 g1 = *(const float4*)(gG0 + 4);
            cvt_store8(as, make_float4(x0.x + g0.x, x0.y + g0.y, x0.z + g0.z, x0.w + g0.w),
                           make_float4(x1.x + g1.x, x1.y + g1.y, x1.z + g1.z, x1.w + g1.w));
        } else {
            *(uint4*)as = make_uint4(0u, 0u, 0u, 0u);
        }
#pragma unroll
        for (int j = 0; j < 4; ++j) {
            const int u = tid + 256 * j;
            const int row = u >> 2, ch = (u & 3) * 8;
            async_b128(Wt + (long)(nbase + row) * Dk + ch, &Bs[0][row][ch]);
        }
    }
    wait_async0();
    __syncthreads();

    for (int i = 0; i < 16; ++i) {
        const int cur = i & 1;
        if (i < 15) {
            const int k1 = (i + 1) * 32;
            __bf16* as = &As[cur ^ 1][srow][scol];
            if (vrow) {
                float4 x0 = *(const float4*)(gX0 + k1);
                float4 x1 = *(const float4*)(gX0 + k1 + 4);
                float4 g0 = *(const float4*)(gG0 + k1);
                float4 g1 = *(const float4*)(gG0 + k1 + 4);
                cvt_store8(as, make_float4(x0.x + g0.x, x0.y + g0.y, x0.z + g0.z, x0.w + g0.w),
                               make_float4(x1.x + g1.x, x1.y + g1.y, x1.z + g1.z, x1.w + g1.w));
            } else {
                *(uint4*)as = make_uint4(0u, 0u, 0u, 0u);
            }
#pragma unroll
            for (int j = 0; j < 4; ++j) {
                const int u = tid + 256 * j;
                const int row = u >> 2, ch = (u & 3) * 8;
                async_b128(Wt + (long)(nbase + row) * Dk + k1 + ch, &Bs[cur ^ 1][row][ch]);
            }
        }
        const int lr = lane & 15;
        const int ka = (lane < 16) ? 0 : 8;
        const int kb = (lane < 16) ? 0 : 8;
        v16bf a[2], b[4];
#pragma unroll
        for (int mt = 0; mt < 2; ++mt)
            a[mt] = load_frag(&As[cur][wM * 32 + mt * 16 + lr][ka],
                              &As[cur][wM * 32 + mt * 16 + lr][ka + 16]);
#pragma unroll
        for (int nt = 0; nt < 4; ++nt)
            b[nt] = load_frag(&Bs[cur][wN * 64 + nt * 16 + lr][kb],
                              &Bs[cur][wN * 64 + nt * 16 + lr][kb + 16]);
#pragma unroll
        for (int mt = 0; mt < 2; ++mt)
#pragma unroll
            for (int nt = 0; nt < 4; ++nt)
                c[mt][nt] = wmma_bf16(a[mt], b[nt], c[mt][nt]);
        wait_async0();
        __syncthreads();
    }

    const int lc      = lane & 15;
    const int colbase = nbase + wN * 64;
#pragma unroll
    for (int mt = 0; mt < 2; ++mt) {
#pragma unroll
        for (int r = 0; r < 8; ++r) {
            const int node = mbase + wM * 32 + mt * 16 + ((lane >> 4) << 3) + r;
            if (node < NNODES) {
#pragma unroll
                for (int nt = 0; nt < 4; ++nt) {
                    const int col = colbase + nt * 16 + lc;
                    float h = fmaxf(c[mt][nt][r] + bv[col], 0.0f);
                    H1[(long)node * Dk + col] = (__bf16)h;
                }
            }
        }
    }
}

// ------------------------------------------- node GEMM 2 + BN partial stats
// h2 = H1 @ W2 + b2 ; out<-h2 ; sum/sumsq per feature via LDS reduction
__global__ __launch_bounds__(256)
void node_gemm2(const __bf16* __restrict__ H1, const __bf16* __restrict__ Wt,
                const float* __restrict__ bv, float* __restrict__ out,
                float* __restrict__ gsum, float* __restrict__ gsq) {
    __shared__ __bf16 As[2][64][32];
    __shared__ __bf16 Bs[2][256][32];
    __shared__ float  colsum[256];
    __shared__ float  colsq[256];

    const int tid  = threadIdx.x;
    const int lane = tid & 31;
    const int wave = tid >> 5;
    const int wM   = wave & 1;
    const int wN   = wave >> 1;
    const int mbase = blockIdx.x * 64;
    const int nbase = blockIdx.y * 256;

    const int srow = tid >> 2;
    const int scol = (tid & 3) * 8;
    // clamp row so async reads stay in-bounds; garbage rows are discarded below
    const long nodeC = (mbase + srow < NNODES) ? (long)(mbase + srow) : (long)(NNODES - 1);

    v8f c[2][4] = {};

    const __bf16* gA0 = H1 + nodeC * Dk + scol;

    {
        async_b128(gA0, &As[0][srow][scol]);
#pragma unroll
        for (int j = 0; j < 4; ++j) {
            const int u = tid + 256 * j;
            const int row = u >> 2, ch = (u & 3) * 8;
            async_b128(Wt + (long)(nbase + row) * Dk + ch, &Bs[0][row][ch]);
        }
    }
    wait_async0();
    __syncthreads();

    for (int i = 0; i < 16; ++i) {
        const int cur = i & 1;
        if (i < 15) {
            const int k1 = (i + 1) * 32;
            async_b128(gA0 + k1, &As[cur ^ 1][srow][scol]);
#pragma unroll
            for (int j = 0; j < 4; ++j) {
                const int u = tid + 256 * j;
                const int row = u >> 2, ch = (u & 3) * 8;
                async_b128(Wt + (long)(nbase + row) * Dk + k1 + ch, &Bs[cur ^ 1][row][ch]);
            }
        }
        const int lr = lane & 15;
        const int ka = (lane < 16) ? 0 : 8;
        const int kb = (lane < 16) ? 0 : 8;
        v16bf a[2], b[4];
#pragma unroll
        for (int mt = 0; mt < 2; ++mt)
            a[mt] = load_frag(&As[cur][wM * 32 + mt * 16 + lr][ka],
                              &As[cur][wM * 32 + mt * 16 + lr][ka + 16]);
#pragma unroll
        for (int nt = 0; nt < 4; ++nt)
            b[nt] = load_frag(&Bs[cur][wN * 64 + nt * 16 + lr][kb],
                              &Bs[cur][wN * 64 + nt * 16 + lr][kb + 16]);
#pragma unroll
        for (int mt = 0; mt < 2; ++mt)
#pragma unroll
            for (int nt = 0; nt < 4; ++nt)
                c[mt][nt] = wmma_bf16(a[mt], b[nt], c[mt][nt]);
        wait_async0();
        __syncthreads();
    }

    colsum[tid] = 0.0f;
    colsq[tid]  = 0.0f;
    __syncthreads();

    const int lc      = lane & 15;
    const int colbase = nbase + wN * 64;
#pragma unroll
    for (int nt = 0; nt < 4; ++nt) {
        const int lcol = wN * 64 + nt * 16 + lc;
        const int col  = colbase + nt * 16 + lc;
        const float bias = bv[col];
        float ps = 0.0f, pq = 0.0f;
#pragma unroll
        for (int mt = 0; mt < 2; ++mt) {
#pragma unroll
            for (int r = 0; r < 8; ++r) {
                const int node = mbase + wM * 32 + mt * 16 + ((lane >> 4) << 3) + r;
                if (node < NNODES) {
                    float v = c[mt][nt][r] + bias;
                    out[(long)node * Dk + col] = v;
                    ps += v; pq += v * v;
                }
            }
        }
        atomicAdd(&colsum[lcol], ps);   // ds_add_f32
        atomicAdd(&colsq[lcol],  pq);
    }
    __syncthreads();
    atomicAdd(&gsum[nbase + tid], colsum[tid]);
    atomicAdd(&gsq[nbase + tid],  colsq[tid]);
}

// ----------------------------------------------------------------- BatchNorm
__global__ __launch_bounds__(256)
void bn_stats(const float* __restrict__ gsum, const float* __restrict__ gsq,
              const float* __restrict__ gamma, const float* __restrict__ beta,
              float* __restrict__ scale, float* __restrict__ shift) {
    const int c = blockIdx.x * 256 + threadIdx.x;
    if (c < Dk) {
        const float invN = 1.0f / (float)NNODES;
        float mean = gsum[c] * invN;
        float var  = gsq[c] * invN - mean * mean;   // biased variance
        float s    = rsqrtf(var + 1e-5f) * gamma[c];
        scale[c] = s;
        shift[c] = beta[c] - mean * s;
    }
}

__global__ __launch_bounds__(256)
void bn_apply(float* __restrict__ out, const float* __restrict__ scale,
              const float* __restrict__ shift) {
    const long i  = ((long)blockIdx.x * 256 + threadIdx.x) * 4;
    const int col = (int)(i & (Dk - 1));
    float4 v = *(float4*)(out + i);
    float4 s = *(const float4*)(scale + col);
    float4 b = *(const float4*)(shift + col);
    v.x = v.x * s.x + b.x;
    v.y = v.y * s.y + b.y;
    v.z = v.z * s.z + b.z;
    v.w = v.w * s.w + b.w;
    *(float4*)(out + i) = v;
}

// ------------------------------------------------------------------- launch
extern "C" void kernel_launch(void* const* d_in, const int* in_sizes, int n_in,
                              void* d_out, int out_size, void* d_ws, size_t ws_size,
                              hipStream_t stream) {
    const float*     x     = (const float*)d_in[0];
    const long long* eidx  = (const long long*)d_in[1];
    const float*     ea    = (const float*)d_in[2];
    const float*     W_e   = (const float*)d_in[3];
    const float*     b_e   = (const float*)d_in[4];
    const float*     W1    = (const float*)d_in[5];
    const float*     b1v   = (const float*)d_in[6];
    const float*     W2    = (const float*)d_in[7];
    const float*     b2v   = (const float*)d_in[8];
    const float*     gamma = (const float*)d_in[9];
    const float*     beta  = (const float*)d_in[10];
    float*           out   = (float*)d_out;

    // workspace carve-up (aggr first so zero-init is one contiguous pass)
    float* aggr  = (float*)d_ws;                    // N*D f32        (51.2 MB)
    float* gsum  = aggr + (long)NNODES * Dk;        // 512 f32
    float* gsq   = gsum + Dk;                       // 512 f32
    float* scale = gsq + Dk;                        // 512 f32
    float* shift = scale + Dk;                      // 512 f32
    __bf16* Wte  = (__bf16*)(shift + Dk);           // 512*512 bf16
    __bf16* Wt1  = Wte + (long)Dk * Dk;
    __bf16* Wt2  = Wt1 + (long)Dk * Dk;
    __bf16* H1   = Wt2 + (long)Dk * Dk;             // N*D bf16       (25.6 MB)

    // 1) zero aggr + sums:  (N*D + 1024) floats = 12,801,024 -> 3,200,256 float4
    zero_ws_kernel<<<12501, 256, 0, stream>>>((float4*)aggr);

    // 2) weights -> bf16, column-major
    cast_transpose_kernel<<<Dk, 256, 0, stream>>>(W_e, Wte);
    cast_transpose_kernel<<<Dk, 256, 0, stream>>>(W1,  Wt1);
    cast_transpose_kernel<<<Dk, 256, 0, stream>>>(W2,  Wt2);

    // 3) edge GEMM + gather + scatter-add   (400000/64 = 6250 M-tiles, 2 N-tiles)
    edge_gemm_scatter<<<dim3(NEDGES / 64, Dk / 256), 256, 0, stream>>>(
        ea, Wte, b_e, x, eidx, aggr);

    // 4) node MLP layer 1 (ReLU, bf16 out)
    node_gemm1<<<dim3((NNODES + 63) / 64, Dk / 256), 256, 0, stream>>>(
        x, aggr, Wt1, b1v, H1);

    // 5) node MLP layer 2 + BN partial sums
    node_gemm2<<<dim3((NNODES + 63) / 64, Dk / 256), 256, 0, stream>>>(
        H1, Wt2, b2v, out, gsum, gsq);

    // 6) BN finalize
    bn_stats<<<2, 256, 0, stream>>>(gsum, gsq, gamma, beta, scale, shift);
    bn_apply<<<(NNODES * Dk / 4) / 256, 256, 0, stream>>>(out, scale, shift);
}
